// NoLSTM_29411936043391
// MI455X (gfx1250) — compile-verified, matrix operand order checked by reading, not run on hardware
//
#include <hip/hip_runtime.h>

// Problem constants (match reference)
#define NN   2048
#define DIN  8
#define DC   8
#define DH   32
#define KITER 2

typedef __attribute__((ext_vector_type(16))) _Float16 v16h;
typedef __attribute__((ext_vector_type(8)))  float    v8f;

// K/channel index owned by (vgpr-half idx, lane-group g) per CDNA5 16-bit
// A/B WMMA layout: lanes 0-15 hold K in {0..7, 16..23}, lanes 16-31 hold
// K in {8..15, 24..31}; VGPR k holds K = (k%4)*2 (+16 if k>=4), two halves.
__device__ __forceinline__ int chanmap(int idx, int g) {
    int k = idx >> 1, h = idx & 1;
    return ((k & 3) << 1) + h + ((k >> 2) << 4) + (g << 3);
}

// ---------------------------------------------------------------------------
// h = [x | comms] @ W_enc + b_enc        (N, DIN+DC) @ (DIN+DC, DH)
// ---------------------------------------------------------------------------
__global__ void encode_kernel(const float* __restrict__ x,
                              const float* __restrict__ comms,
                              const float* __restrict__ W_enc,
                              const float* __restrict__ b_enc,
                              float* __restrict__ h) {
    int t = blockIdx.x * blockDim.x + threadIdx.x;
    if (t >= NN * DH) return;
    int i = t / DH, c = t % DH;
    float acc = b_enc[c];
#pragma unroll
    for (int d = 0; d < DIN; ++d) acc += x[i * DIN + d] * W_enc[d * DH + c];
#pragma unroll
    for (int d = 0; d < DC; ++d)  acc += comms[i * DC + d] * W_enc[(DIN + d) * DH + c];
    h[t] = acc;
}

// ---------------------------------------------------------------------------
// q/kk/v = h @ W{q,k,v} + b{q,k,v}       (N, DH) @ (DH, DH), tiny GEMMs
// ---------------------------------------------------------------------------
__global__ void qkv_kernel(const float* __restrict__ h,
                           const float* __restrict__ Wq, const float* __restrict__ bq,
                           const float* __restrict__ Wk, const float* __restrict__ bk,
                           const float* __restrict__ Wv, const float* __restrict__ bv,
                           float* __restrict__ q, float* __restrict__ kk,
                           float* __restrict__ v) {
    int t = blockIdx.x * blockDim.x + threadIdx.x;
    if (t >= NN * DH) return;
    int i = t / DH, c = t % DH;
    float aq = bq[c], ak = bk[c], av = bv[c];
#pragma unroll 8
    for (int d = 0; d < DH; ++d) {
        float hd = h[i * DH + d];
        aq += hd * Wq[d * DH + c];
        ak += hd * Wk[d * DH + c];
        av += hd * Wv[d * DH + c];
    }
    q[t] = aq; kk[t] = ak; v[t] = av;
}

// ---------------------------------------------------------------------------
// Fused pairwise-MLP attention, one wave (32 threads) per row i.
//   dots[i,j] = kk[i].q[j];  f1 = relu([dots,adj,dense]@W1+b1)   (VALU, in
//   WMMA-A layout);  f2 = relu(f1@W2+b2)  (v_wmma_f32_16x16x32_f16);
//   logit = f2@W3+b3 (shfl reduce);  online softmax over j;  acc = focus@v.
//   32 j's per loop iteration = 2 independent 16-pair WMMA groups, so the
//   scheduler can overlap the two WMMA/reduce chains and amortize the
//   online-softmax rescale.  hout[i] = acc/l + hin[i].
// ---------------------------------------------------------------------------
__global__ __launch_bounds__(32) void attn_kernel(
    const float* __restrict__ q, const float* __restrict__ kk,
    const float* __restrict__ v, const float* __restrict__ adj,
    const float* __restrict__ dense,
    const float* __restrict__ W1, const float* __restrict__ b1,
    const float* __restrict__ W2, const float* __restrict__ b2,
    const float* __restrict__ W3, const float* __restrict__ b3,
    const float* __restrict__ hin, float* __restrict__ hout) {
    const int i    = blockIdx.x;
    const int lane = threadIdx.x;
    const int g    = lane >> 4;    // lane group (0: lanes 0-15, 1: lanes 16-31)
    const int nl   = lane & 15;

    // Per-lane W1/b1 constants for this lane's 16 K-channels (A layout)
    float w1d[16], w1a[16], w1n[16], b1v[16];
#pragma unroll
    for (int idx = 0; idx < 16; ++idx) {
        int c = chanmap(idx, g);
        w1d[idx] = W1[0 * DH + c];
        w1a[idx] = W1[1 * DH + c];
        w1n[idx] = W1[2 * DH + c];
        b1v[idx] = b1[c];
    }
    // W2 in f16 WMMA-B layout: B0 -> output cols 0..15, B1 -> cols 16..31
    v16h B0, B1;
#pragma unroll
    for (int idx = 0; idx < 16; ++idx) {
        int k = chanmap(idx, g);
        B0[idx] = (_Float16)W2[k * DH + nl];
        B1[idx] = (_Float16)W2[k * DH + nl + 16];
    }
    const float b2n0 = b2[nl], b2n1 = b2[nl + 16];
    const float w3n0 = W3[nl], w3n1 = W3[nl + 16];
    const float b3s  = b3[0];

    // Row i of kk (wave-uniform)
    float kki[DH];
#pragma unroll
    for (int d = 0; d < DH; ++d) kki[d] = kk[i * DH + d];

    float m = -__builtin_inff();   // running max (uniform)
    float l = 0.0f;                // running denom (uniform)
    float acc = 0.0f;              // weighted-V accumulator, channel = lane

    // logits for one 16-pair block starting at jb0; lp[r] = logit[jb0+r+8*g]
    auto block_logits = [&](int jb0, float* lp) {
        const int j = jb0 + nl;    // this lane's pair (both groups mirror)
        // dots[i,j] = kk[i] . q[j]
        float dot = 0.0f;
        const float* qr = q + j * DH;
#pragma unroll
        for (int d = 0; d < DH; ++d) dot += kki[d] * qr[d];
        const float fa = adj[i * NN + j];
        const float fn = dense[i * NN + j];

        // f1 = relu([dot,adj,dense] @ W1 + b1), built directly in A layout
        v16h A;
#pragma unroll
        for (int idx = 0; idx < 16; ++idx) {
            float t = fmaf(dot, w1d[idx],
                      fmaf(fa,  w1a[idx],
                      fmaf(fn,  w1n[idx], b1v[idx])));
            A[idx] = (_Float16)fmaxf(t, 0.0f);
        }

        // f2 = f1 @ W2 (bias/relu applied post-WMMA)
        v8f c0 = {}, c1 = {};
        v8f D0 = __builtin_amdgcn_wmma_f32_16x16x32_f16(
            false, A, false, B0, (short)0, c0, false, false);
        v8f D1 = __builtin_amdgcn_wmma_f32_16x16x32_f16(
            false, A, false, B1, (short)0, c1, false, false);

        // logit[p] = relu(f2 + b2) @ W3 + b3 ; register r holds pair r + 8*g
#pragma unroll
        for (int r = 0; r < 8; ++r) {
            float t = fmaxf(D0[r] + b2n0, 0.0f) * w3n0 +
                      fmaxf(D1[r] + b2n1, 0.0f) * w3n1;
#pragma unroll
            for (int msk = 1; msk < 16; msk <<= 1)
                t += __shfl_xor(t, msk, 32);
            lp[r] = t + b3s;       // uniform within each 16-lane group
        }
    };

    for (int jb = 0; jb < NN; jb += 32) {
        float lpA[8], lpB[8];
        block_logits(jb,      lpA);   // pairs jb    .. jb+15
        block_logits(jb + 16, lpB);   // pairs jb+16 .. jb+31

        // block max across all 32 pairs (both groups)
        float mb = fmaxf(lpA[0], lpB[0]);
#pragma unroll
        for (int r = 1; r < 8; ++r) mb = fmaxf(mb, fmaxf(lpA[r], lpB[r]));
        mb = fmaxf(mb, __shfl_xor(mb, 16, 32));
        const float mnew  = fmaxf(m, mb);
        const float scale = __expf(m - mnew);
        acc *= scale;
        l   *= scale;

        // acc[c] += sum_p exp(logit[p]-mnew) * v[jb+p][c]   (c = lane)
#pragma unroll
        for (int r = 0; r < 8; ++r) {
            float a0 = __shfl(lpA[r], 0, 32);    // pair jb + r
            float a1 = __shfl(lpA[r], 16, 32);   // pair jb + 8 + r
            float bb0 = __shfl(lpB[r], 0, 32);   // pair jb + 16 + r
            float bb1 = __shfl(lpB[r], 16, 32);  // pair jb + 24 + r
            float w0 = __expf(a0 - mnew);
            float w1 = __expf(a1 - mnew);
            float w2 = __expf(bb0 - mnew);
            float w3 = __expf(bb1 - mnew);
            l   += (w0 + w1) + (w2 + w3);
            acc  = fmaf(w0, v[(jb + r)      * DH + lane], acc);
            acc  = fmaf(w1, v[(jb + 8 + r)  * DH + lane], acc);
            acc  = fmaf(w2, v[(jb + 16 + r) * DH + lane], acc);
            acc  = fmaf(w3, v[(jb + 24 + r) * DH + lane], acc);
        }
        m = mnew;
    }

    // residual
    hout[i * DH + lane] = acc / l + hin[i * DH + lane];
}

// ---------------------------------------------------------------------------
// out[i] = h[i] @ W_dec + b_dec, masked to -inf where mask==0
// ---------------------------------------------------------------------------
__global__ void decode_kernel(const float* __restrict__ h,
                              const float* __restrict__ W_dec,
                              const float* __restrict__ b_dec,
                              const float* __restrict__ mask,
                              float* __restrict__ out) {
    int i = blockIdx.x * blockDim.x + threadIdx.x;
    if (i >= NN) return;
    float acc = b_dec[0];
#pragma unroll 8
    for (int c = 0; c < DH; ++c) acc += h[i * DH + c] * W_dec[c];
    out[i] = (mask[i] == 0.0f) ? -__builtin_inff() : acc;
}

// ---------------------------------------------------------------------------
extern "C" void kernel_launch(void* const* d_in, const int* in_sizes, int n_in,
                              void* d_out, int out_size, void* d_ws, size_t ws_size,
                              hipStream_t stream) {
    const float* x     = (const float*)d_in[0];
    const float* comms = (const float*)d_in[1];
    const float* adj   = (const float*)d_in[2];
    const float* dense = (const float*)d_in[3];
    const float* mask  = (const float*)d_in[4];
    const float* W_enc = (const float*)d_in[5];
    const float* b_enc = (const float*)d_in[6];
    const float* Wq    = (const float*)d_in[7];
    const float* bq    = (const float*)d_in[8];
    const float* Wk    = (const float*)d_in[9];
    const float* bk    = (const float*)d_in[10];
    const float* Wv    = (const float*)d_in[11];
    const float* bv    = (const float*)d_in[12];
    const float* W1    = (const float*)d_in[13];
    const float* b1    = (const float*)d_in[14];
    const float* W2    = (const float*)d_in[15];
    const float* b2    = (const float*)d_in[16];
    const float* W3    = (const float*)d_in[17];
    const float* b3    = (const float*)d_in[18];
    const float* W_dec = (const float*)d_in[19];
    const float* b_dec = (const float*)d_in[20];
    float* out = (float*)d_out;

    // workspace: hA | hB | q | kk | v  (each N*DH floats = 256 KB; 1.25 MB total)
    float* ws  = (float*)d_ws;
    float* hA  = ws;
    float* hB  = ws + 1 * NN * DH;
    float* qb  = ws + 2 * NN * DH;
    float* kkb = ws + 3 * NN * DH;
    float* vb  = ws + 4 * NN * DH;

    const int tpb = 256;
    const int nElem = NN * DH;

    encode_kernel<<<(nElem + tpb - 1) / tpb, tpb, 0, stream>>>(
        x, comms, W_enc, b_enc, hA);

    for (int it = 0; it < KITER; ++it) {
        const float* hi = (it & 1) ? hB : hA;
        float*       ho = (it & 1) ? hA : hB;
        qkv_kernel<<<(nElem + tpb - 1) / tpb, tpb, 0, stream>>>(
            hi, Wq, bq, Wk, bk, Wv, bv, qb, kkb, vb);
        attn_kernel<<<NN, 32, 0, stream>>>(
            qb, kkb, vb, adj, dense, W1, b1, W2, b2, W3, b3, hi, ho);
    }

    // KITER == 2 -> final h is in hA
    decode_kernel<<<(NN + tpb - 1) / tpb, tpb, 0, stream>>>(
        hA, W_dec, b_dec, mask, out);
}
